// MultiHeadAttention_42726334660755
// MI455X (gfx1250) — matrix operand order repaired and checked by reading, not compiled, not run on hardware
//
#include <hip/hip_runtime.h>

// ---------------------------------------------------------------------------
// MI455X (gfx1250) multi-head attention, bf16 WMMA with f32 accumulation,
// async global->LDS staging with double buffering in the GEMMs.
// B=4, S=2048, D=1024, H=16, Dh=64.
// ---------------------------------------------------------------------------

typedef __attribute__((ext_vector_type(16))) __bf16 v16bf;
typedef __attribute__((ext_vector_type(8)))  float  v8f;
typedef __attribute__((ext_vector_type(4)))  int    v4i;

union FragBF { v16bf v; unsigned int u[8]; };
union FragF  { v8f   v; float        f[8]; };

#if __has_builtin(__builtin_amdgcn_global_load_async_to_lds_b128)
#define HAVE_ASYNC_LDS 1
#endif

// Copy 16 bytes global -> LDS. Async (ASYNCcnt) when available.
__device__ __forceinline__ void async_ld_b128(const void* g, void* l) {
#if defined(HAVE_ASYNC_LDS)
  __builtin_amdgcn_global_load_async_to_lds_b128(
      (__attribute__((address_space(1))) v4i*)(g),
      (__attribute__((address_space(3))) v4i*)(l), 0, 0);
#else
  *(uint4*)l = *(const uint4*)g;
#endif
}

__device__ __forceinline__ void wait_async0() {
#if __has_builtin(__builtin_amdgcn_s_wait_asynccnt)
  __builtin_amdgcn_s_wait_asynccnt(0);
#else
  asm volatile("s_wait_asynccnt 0" ::: "memory");
#endif
}

__device__ __forceinline__ unsigned short f2bf(float f) {
  unsigned int x = __float_as_uint(f);
  unsigned int r = x + 0x7FFFu + ((x >> 16) & 1u);   // round-to-nearest-even
  return (unsigned short)(r >> 16);
}

// A-matrix 16x32 bf16 fragment: lane&15 = M row; per half:
//   u[0..7] = K {half*16+0 .. half*16+15} (contiguous, matches B fragment)
__device__ __forceinline__ void load_afrag(FragBF& f, const unsigned int* rp,
                                           int half, int cbase) {
#pragma unroll
  for (int i = 0; i < 8; ++i) f.u[i] = rp[cbase + half * 8 + i];
}

// B-matrix 32x16 bf16 fragment: lane&15 = N column; lanes 0-15 hold K=0..15,
// lanes 16-31 hold K=16..31 (contiguous). rp points at [column][k] row in LDS.
__device__ __forceinline__ void load_bfrag(FragBF& f, const unsigned int* rp,
                                           int half, int cbase) {
#pragma unroll
  for (int i = 0; i < 8; ++i) f.u[i] = rp[cbase + half * 8 + i];
}

__device__ __forceinline__ v8f wmma_bf16(const FragBF& a, const FragBF& b, v8f c) {
  return __builtin_amdgcn_wmma_f32_16x16x32_bf16(false, a.v, false, b.v,
                                                 (short)0, c, false, false);
}

// ---------------------------------------------------------------------------
// f32 -> bf16 cast
// ---------------------------------------------------------------------------
__global__ void cast_f32_bf16(const float* __restrict__ in,
                              unsigned short* __restrict__ out, int n) {
  int i = blockIdx.x * blockDim.x + threadIdx.x;
  if (i < n) out[i] = f2bf(in[i]);
}

// ---------------------------------------------------------------------------
// GEMM: C[M,N] = A[M,K](bf16) * B[K,N](bf16), f32 accumulate.
// Block tile 128x64, 256 threads (8 waves), wave tile 32x32 (2x2 WMMA).
// Double-buffered LDS, async global->LDS staging for the A tile.
// mode 0: outF[gm*N+gn] = c + bias[gn]           (f32, row-major)
// mode 1: outB head-major bf16 [B,H,S,64] from (gm -> b,s ; gn -> h,d)
// ---------------------------------------------------------------------------
__global__ __launch_bounds__(256) void gemm_bf16_k(
    const unsigned short* __restrict__ A,
    const unsigned short* __restrict__ Bm,
    float* __restrict__ outF,
    unsigned short* __restrict__ outB,
    const float* __restrict__ bias,
    int M, int N, int K, int mode) {
  __shared__ unsigned short As[2][128][40];  // [buf][m][k], 80B row stride
  __shared__ unsigned short Bs[2][64][40];   // [buf][n][k] (transposed store)

  const int tid  = threadIdx.x;
  const int lane = tid & 31, lr = lane & 15, half = lane >> 4;
  const int wid  = tid >> 5;
  const int wm   = wid >> 1, wn = wid & 1;
  const int m0   = blockIdx.y * 128;
  const int n0   = blockIdx.x * 64;

  FragF c[2][2];
#pragma unroll
  for (int tm = 0; tm < 2; ++tm)
#pragma unroll
    for (int tn = 0; tn < 2; ++tn)
      c[tm][tn].v = (v8f){0.f, 0.f, 0.f, 0.f, 0.f, 0.f, 0.f, 0.f};

  const int arow = tid >> 1, acol = (tid & 1) * 16;  // 128 rows x 32 cols
  const int brow = tid >> 3, bcol = (tid & 7) * 8;   // 32 rows x 64 cols

  // Stage one 128x32 A tile (async) and one 32x64 B tile (transposed) to LDS.
  auto stage = [&](int kt, int buf) {
    const unsigned short* ag = A + (size_t)(m0 + arow) * K + kt + acol;
    async_ld_b128(ag,     &As[buf][arow][acol]);
    async_ld_b128(ag + 8, &As[buf][arow][acol + 8]);
    const uint4 b0 = *(const uint4*)(Bm + (size_t)(kt + brow) * N + n0 + bcol);
    unsigned int w[4] = {b0.x, b0.y, b0.z, b0.w};
#pragma unroll
    for (int q = 0; q < 4; ++q) {
      Bs[buf][bcol + q * 2 + 0][brow] = (unsigned short)(w[q] & 0xffff);
      Bs[buf][bcol + q * 2 + 1][brow] = (unsigned short)(w[q] >> 16);
    }
  };

  stage(0, 0);
  const int nkt = K >> 5;
  for (int it = 0; it < nkt; ++it) {
    const int buf = it & 1;
    wait_async0();        // own async loads for this buffer done
    __syncthreads();      // everyone's staging for this buffer done
    if (it + 1 < nkt) stage((it + 1) << 5, buf ^ 1);

    FragBF aF[2], bF[2];
#pragma unroll
    for (int tm = 0; tm < 2; ++tm)
      load_afrag(aF[tm], (const unsigned int*)&As[buf][wm * 32 + tm * 16 + lr][0],
                 half, 0);
#pragma unroll
    for (int tn = 0; tn < 2; ++tn)
      load_bfrag(bF[tn], (const unsigned int*)&Bs[buf][wn * 32 + tn * 16 + lr][0],
                 half, 0);
#pragma unroll
    for (int tm = 0; tm < 2; ++tm)
#pragma unroll
      for (int tn = 0; tn < 2; ++tn)
        c[tm][tn].v = wmma_bf16(aF[tm], bF[tn], c[tm][tn].v);
    __syncthreads();      // all reads of this buffer done before it is reused
  }

#pragma unroll
  for (int tm = 0; tm < 2; ++tm)
#pragma unroll
    for (int tn = 0; tn < 2; ++tn)
#pragma unroll
      for (int j = 0; j < 8; ++j) {
        int gm = m0 + wm * 32 + tm * 16 + half * 8 + j;
        int gn = n0 + wn * 32 + tn * 16 + lr;
        float v = c[tm][tn].f[j];
        if (mode == 0) {
          outF[(size_t)gm * N + gn] = v + bias[gn];
        } else {
          int b = gm >> 11, s = gm & 2047;      // S = 2048
          int h = gn >> 6,  d = gn & 63;        // Dh = 64
          outB[(((size_t)(b * 16 + h)) * 2048 + s) * 64 + d] = f2bf(v);
        }
      }
}

// ---------------------------------------------------------------------------
// Flash attention: grid (B*H, S/64), 128 threads (4 waves x 16 query rows).
// Q,K,V bf16 head-major [B,H,S,64]; ctx bf16 [B*S, 1024].
// ---------------------------------------------------------------------------
__global__ __launch_bounds__(128) void attn_k(
    const unsigned short* __restrict__ Qm,
    const unsigned short* __restrict__ Km,
    const unsigned short* __restrict__ Vm,
    unsigned short* __restrict__ ctx) {
  __shared__ unsigned short Ks[64][72];       // [key][d], 144B row stride
  __shared__ unsigned short Vs[64][72];       // [d][key] (transposed)
  __shared__ unsigned short Ps[4][16][72];    // per-wave P tile (bf16)

  const int tid  = threadIdx.x;
  const int lane = tid & 31, lr = lane & 15, half = lane >> 4;
  const int wid  = tid >> 5;
  const int bh   = blockIdx.x;                // b*16 + h
  const int qt   = blockIdx.y;                // q tile (64 rows)
  const int b    = bh >> 4, h = bh & 15;
  const size_t base = (size_t)bh * 2048 * 64;
  const int q0   = qt * 64 + wid * 16;

  // Q fragments (16 rows x 64 dh), 2 K-chunks, loaded straight from global.
  FragBF aQ[2];
  {
    const unsigned int* qp = (const unsigned int*)(Qm + base + (size_t)(q0 + lr) * 64);
#pragma unroll
    for (int kc = 0; kc < 2; ++kc) load_afrag(aQ[kc], qp, half, kc * 16);
  }

  float Mrow[8], Lrow[8];
  FragF o[4];
#pragma unroll
  for (int j = 0; j < 8; ++j) { Mrow[j] = -3.0e38f; Lrow[j] = 0.f; }
#pragma unroll
  for (int nb = 0; nb < 4; ++nb)
    o[nb].v = (v8f){0.f, 0.f, 0.f, 0.f, 0.f, 0.f, 0.f, 0.f};

  const int krow = tid >> 1, ch = (tid & 1) * 32;   // tile-load mapping
  const int qrow_base = qt * 64 + wid * 16 + half * 8;

  for (int kt = 0; kt <= qt; ++kt) {
    const int kbase = kt * 64;
    // K tile -> LDS [key][d] (async row copy, 4 x b128 per thread)
    {
      const unsigned short* kg = Km + base + (size_t)(kbase + krow) * 64 + ch;
      unsigned short* kl = &Ks[krow][ch];
      async_ld_b128(kg,      kl);
      async_ld_b128(kg + 8,  kl + 8);
      async_ld_b128(kg + 16, kl + 16);
      async_ld_b128(kg + 24, kl + 24);
    }
    // V tile -> LDS transposed [d][key]
    {
      const uint4* vp = (const uint4*)(Vm + base + (size_t)(kbase + krow) * 64 + ch);
#pragma unroll
      for (int i = 0; i < 4; ++i) {
        uint4 t = vp[i];
        unsigned int w[4] = {t.x, t.y, t.z, t.w};
#pragma unroll
        for (int q = 0; q < 4; ++q) {
          Vs[ch + i * 8 + q * 2 + 0][krow] = (unsigned short)(w[q] & 0xffff);
          Vs[ch + i * 8 + q * 2 + 1][krow] = (unsigned short)(w[q] >> 16);
        }
      }
    }
    if (kt < qt)  // prefetch next K tile line into GL2
      __builtin_prefetch(Km + base + (size_t)(kbase + 64 + krow) * 64 + ch, 0, 1);
    wait_async0();
    __syncthreads();

    // scores: S = Q (16x64) * K_tile^T (64x64) -> 4 column blocks of 16
    FragF sc[4];
#pragma unroll
    for (int nb = 0; nb < 4; ++nb) {
      sc[nb].v = (v8f){0.f, 0.f, 0.f, 0.f, 0.f, 0.f, 0.f, 0.f};
      const unsigned int* rp = (const unsigned int*)&Ks[nb * 16 + lr][0];
      FragBF kF;
#pragma unroll
      for (int kc = 0; kc < 2; ++kc) {
        load_bfrag(kF, rp, half, kc * 16);
        sc[nb].v = wmma_bf16(aQ[kc], kF, sc[nb].v);
      }
    }

    // scale + causal mask + row max (half-wave shuffle reduce)
    float rmax[8];
#pragma unroll
    for (int j = 0; j < 8; ++j) rmax[j] = -3.0e38f;
#pragma unroll
    for (int nb = 0; nb < 4; ++nb) {
      int kcol = kbase + nb * 16 + lr;
#pragma unroll
      for (int j = 0; j < 8; ++j) {
        float s = sc[nb].f[j] * 0.125f;            // 1/sqrt(64)
        if (kcol > qrow_base + j) s = -3.0e38f;    // causal
        sc[nb].f[j] = s;
        rmax[j] = fmaxf(rmax[j], s);
      }
    }
#pragma unroll
    for (int m = 1; m < 16; m <<= 1)
#pragma unroll
      for (int j = 0; j < 8; ++j)
        rmax[j] = fmaxf(rmax[j], __shfl_xor(rmax[j], m, 32));

    float alpha[8];
#pragma unroll
    for (int j = 0; j < 8; ++j) {
      float mn = fmaxf(Mrow[j], rmax[j]);
      alpha[j] = __expf(Mrow[j] - mn);
      Mrow[j]  = mn;
    }

    // P = exp(s - M), row sums, rescale O, stage P to LDS as bf16
    float rsum[8];
#pragma unroll
    for (int j = 0; j < 8; ++j) rsum[j] = 0.f;
#pragma unroll
    for (int nb = 0; nb < 4; ++nb)
#pragma unroll
      for (int j = 0; j < 8; ++j) {
        float p = __expf(sc[nb].f[j] - Mrow[j]);
        sc[nb].f[j] = p;
        rsum[j] += p;
        o[nb].f[j] *= alpha[j];
        Ps[wid][half * 8 + j][nb * 16 + lr] = f2bf(p);
      }
#pragma unroll
    for (int m = 1; m < 16; m <<= 1)
#pragma unroll
      for (int j = 0; j < 8; ++j) rsum[j] += __shfl_xor(rsum[j], m, 32);
#pragma unroll
    for (int j = 0; j < 8; ++j) Lrow[j] = Lrow[j] * alpha[j] + rsum[j];

    asm volatile("s_wait_dscnt 0" ::: "memory");   // P stores visible to reloads

    // O += P (16x64 keys) * V_tile (64 keys x 64 d)
    FragBF pF[2];
    {
      const unsigned int* rp = (const unsigned int*)&Ps[wid][lr][0];
#pragma unroll
      for (int kc = 0; kc < 2; ++kc) load_afrag(pF[kc], rp, half, kc * 16);
    }
#pragma unroll
    for (int nb = 0; nb < 4; ++nb) {
      const unsigned int* vp = (const unsigned int*)&Vs[nb * 16 + lr][0];
      FragBF vF;
#pragma unroll
      for (int kc = 0; kc < 2; ++kc) {
        load_bfrag(vF, vp, half, kc * 16);
        o[nb].v = wmma_bf16(pF[kc], vF, o[nb].v);
      }
    }
    __syncthreads();
  }

  // normalize and write ctx bf16 [B*S, 1024]
  float inv[8];
#pragma unroll
  for (int j = 0; j < 8; ++j) inv[j] = 1.0f / Lrow[j];
#pragma unroll
  for (int nb = 0; nb < 4; ++nb)
#pragma unroll
    for (int j = 0; j < 8; ++j) {
      int s = qrow_base + j;
      ctx[((size_t)(b * 2048 + s)) * 1024 + h * 64 + nb * 16 + lr] =
          f2bf(o[nb].f[j] * inv[j]);
    }
}

// ---------------------------------------------------------------------------
// Host launcher
// ---------------------------------------------------------------------------
extern "C" void kernel_launch(void* const* d_in, const int* in_sizes, int n_in,
                              void* d_out, int out_size, void* d_ws, size_t ws_size,
                              hipStream_t stream) {
  const float* x  = (const float*)d_in[0];
  const float* Wq = (const float*)d_in[1];
  const float* Wk = (const float*)d_in[2];
  const float* Wv = (const float*)d_in[3];
  const float* Wo = (const float*)d_in[4];
  const float* bo = (const float*)d_in[5];
  float* out = (float*)d_out;

  unsigned short* ws = (unsigned short*)d_ws;
  const size_t XB = (size_t)8192 * 1024;        // x / ctx elems
  const size_t WB = (size_t)1024 * 1024;        // weight elems
  const size_t HB = (size_t)4 * 16 * 2048 * 64; // head-major elems
  unsigned short* xb  = ws;
  unsigned short* wqb = xb + XB;
  unsigned short* wkb = wqb + WB;
  unsigned short* wvb = wkb + WB;
  unsigned short* wob = wvb + WB;
  unsigned short* Qh  = wob + WB;
  unsigned short* Kh  = Qh + HB;
  unsigned short* Vh  = Kh + HB;
  unsigned short* ctx = Vh + HB;

  const int nx = (int)XB, nw = (int)WB;
  cast_f32_bf16<<<(nx + 255) / 256, 256, 0, stream>>>(x,  xb,  nx);
  cast_f32_bf16<<<(nw + 255) / 256, 256, 0, stream>>>(Wq, wqb, nw);
  cast_f32_bf16<<<(nw + 255) / 256, 256, 0, stream>>>(Wk, wkb, nw);
  cast_f32_bf16<<<(nw + 255) / 256, 256, 0, stream>>>(Wv, wvb, nw);
  cast_f32_bf16<<<(nw + 255) / 256, 256, 0, stream>>>(Wo, wob, nw);

  dim3 ggrid(16, 64);  // N/64 x M/128
  gemm_bf16_k<<<ggrid, 256, 0, stream>>>(xb, wqb, nullptr, Qh, nullptr,
                                         8192, 1024, 1024, 1);
  gemm_bf16_k<<<ggrid, 256, 0, stream>>>(xb, wkb, nullptr, Kh, nullptr,
                                         8192, 1024, 1024, 1);
  gemm_bf16_k<<<ggrid, 256, 0, stream>>>(xb, wvb, nullptr, Vh, nullptr,
                                         8192, 1024, 1024, 1);

  attn_k<<<dim3(64, 32), 128, 0, stream>>>(Qh, Kh, Vh, ctx);

  gemm_bf16_k<<<ggrid, 256, 0, stream>>>(ctx, wob, out, nullptr, bo,
                                         8192, 1024, 1024, 0);
}
